// CementPINN_46084999086149
// MI455X (gfx1250) — matrix operand order repaired and checked by reading, not compiled, not run on hardware
//
#include <hip/hip_runtime.h>
#include <hip/hip_bf16.h>
#include <math.h>

// ---------------------------------------------------------------------------
// CementPINN fused MLP for gfx1250 (MI455X).
//   raw = relu(x@W1+b1) -> relu(@W2+b2) -> relu(@W3+b3) -> @W4+b4
//   out = hard_constraints(x, raw)
// Strategy: f16 WMMA (v_wmma_f32_16x16x32_f16) with fp32 accumulation,
// fully fused per 64-row tile, activations ping-ponged in LDS (2x128KB,
// only possible with CDNA5's 320KB WGP LDS). Weights pre-converted to f16
// and transposed to N-major so B fragments load as contiguous b128s.
// ---------------------------------------------------------------------------

typedef __attribute__((ext_vector_type(16))) _Float16 v16h;
typedef __attribute__((ext_vector_type(8)))  _Float16 v8h;
typedef __attribute__((ext_vector_type(8)))  float    v8f;

union Frag16 { v16h v; v8h h[2]; };

#define HDIM   1024
#define ROWS   64            // rows per workgroup
#define NTHR   256           // 8 waves (wave32)

// ---- LDS layout (dynamic shared) ------------------------------------------
// [0, 4096)                 : x tile, 64 x 32 f16 (K padded 8 -> 32)
// [4096, 135168)            : h buffer A, 64 x 1024 f16
// [135168, 266240)          : h buffer B, 64 x 1024 f16
// [266240, 267264)          : 64 x 4 f32 reduction scratch
#define SMEM_XF    0
#define SMEM_H0    4096
#define SMEM_H1    (4096 + 131072)
#define SMEM_RED   (4096 + 262144)
#define SMEM_TOTAL (4096 + 262144 + 1024)

// ---------------------------------------------------------------------------
// Fragment loaders (ISA 7.12.2 layouts, wave32).
// A (16x32 f16): lane L holds row M=L&15; halves 0..7  = K (L<16 ? 0..7  : 8..15)
//                                        halves 8..15 = K (L<16 ?16..23 :24..31)
// -> two 16B contiguous LDS loads per fragment.
__device__ __forceinline__ v16h load_a_frag(const _Float16* __restrict__ src,
                                            int strideH, int row0, int kbase, int lane) {
    const int m  = lane & 15;
    const int hi = lane >> 4;
    const _Float16* p = src + (row0 + m) * strideH + kbase + hi * 8;
    Frag16 f;
    f.h[0] = *(const v8h*)(p);        // K: kbase + hi*8 + 0..7
    f.h[1] = *(const v8h*)(p + 16);   // K: kbase + 16 + hi*8 + 0..7
    return f.v;
}

// B (32x16 f16): lane L holds col N=L&15; halves 0..7 = K (L<16 ? 0..7 : 8..15), halves 8..15 = K (L<16 ? 16..23 : 24..31)
// Weights stored N-major (Wt[n*K + k]) -> two 16-byte runs per lane, 32 bytes apart:
// two global_load_b128.
__device__ __forceinline__ v16h load_b_frag(const _Float16* __restrict__ Wt,
                                            int strideK, int n0, int kbase, int lane) {
    const int n  = lane & 15;
    const int hi = lane >> 4;
    const _Float16* p = Wt + (size_t)(n0 + n) * strideK + kbase + hi * 8;
    Frag16 f;
    f.h[0] = *(const v8h*)(p);
    f.h[1] = *(const v8h*)(p + 16);
    return f.v;
}

// ---------------------------------------------------------------------------
// One dense layer: hout[64 x 1024] = relu(hin[64 x K] @ W[K x 1024] + b)
// hin in LDS (stride == K halves), W transposed f16 in global, hout in LDS.
// Wave tiling: 8 waves = 2 row-groups x 4 col-groups; each wave owns a
// 32x64 patch per N-chunk (2x4 WMMA tiles, 64 f32 acc VGPRs).
__device__ __forceinline__ void layer_wmma(const _Float16* __restrict__ hin, int K,
                                           const _Float16* __restrict__ Wt,
                                           const float* __restrict__ bias,
                                           _Float16* __restrict__ hout) {
    const int lane = threadIdx.x & 31;
    const int wid  = threadIdx.x >> 5;   // 0..7
    const int rowG = wid & 1;            // 2 row groups
    const int colG = wid >> 1;           // 4 col groups
    const int row0 = rowG * 32;
    const int hi   = lane >> 4;
    const int nlo  = lane & 15;

    for (int no = 0; no < HDIM; no += 256) {
        const int n0w = no + colG * 64;

        v8f acc[2][4];
        const v8f vzero = {0.f, 0.f, 0.f, 0.f, 0.f, 0.f, 0.f, 0.f};
        #pragma unroll
        for (int m = 0; m < 2; ++m)
            #pragma unroll
            for (int n = 0; n < 4; ++n)
                acc[m][n] = vzero;

        for (int kb = 0; kb < K; kb += 32) {
            v16h a0 = load_a_frag(hin, K, row0,      kb, lane);
            v16h a1 = load_a_frag(hin, K, row0 + 16, kb, lane);
            v16h b0 = load_b_frag(Wt, K, n0w,      kb, lane);
            v16h b1 = load_b_frag(Wt, K, n0w + 16, kb, lane);
            v16h b2 = load_b_frag(Wt, K, n0w + 32, kb, lane);
            v16h b3 = load_b_frag(Wt, K, n0w + 48, kb, lane);

            acc[0][0] = __builtin_amdgcn_wmma_f32_16x16x32_f16(false, a0, false, b0, (short)0, acc[0][0], false, false);
            acc[0][1] = __builtin_amdgcn_wmma_f32_16x16x32_f16(false, a0, false, b1, (short)0, acc[0][1], false, false);
            acc[0][2] = __builtin_amdgcn_wmma_f32_16x16x32_f16(false, a0, false, b2, (short)0, acc[0][2], false, false);
            acc[0][3] = __builtin_amdgcn_wmma_f32_16x16x32_f16(false, a0, false, b3, (short)0, acc[0][3], false, false);
            acc[1][0] = __builtin_amdgcn_wmma_f32_16x16x32_f16(false, a1, false, b0, (short)0, acc[1][0], false, false);
            acc[1][1] = __builtin_amdgcn_wmma_f32_16x16x32_f16(false, a1, false, b1, (short)0, acc[1][1], false, false);
            acc[1][2] = __builtin_amdgcn_wmma_f32_16x16x32_f16(false, a1, false, b2, (short)0, acc[1][2], false, false);
            acc[1][3] = __builtin_amdgcn_wmma_f32_16x16x32_f16(false, a1, false, b3, (short)0, acc[1][3], false, false);
        }

        // Epilogue: bias + ReLU + f16 convert into LDS.
        // C/D layout: VGPR r, lanes0-15 -> M=r, lanes16-31 -> M=r+8; N = lane&15.
        #pragma unroll
        for (int m = 0; m < 2; ++m) {
            #pragma unroll
            for (int n = 0; n < 4; ++n) {
                const int ncol = n0w + n * 16 + nlo;
                const float bn = bias[ncol];
                #pragma unroll
                for (int r = 0; r < 8; ++r) {
                    const int mrow = row0 + m * 16 + hi * 8 + r;
                    float v = acc[m][n][r] + bn;
                    v = v > 0.f ? v : 0.f;
                    hout[mrow * HDIM + ncol] = (_Float16)v;
                }
            }
        }
    }
}

// ---------------------------------------------------------------------------
// Weight prep: f32 -> f16, transpose to N-major.
__global__ __launch_bounds__(256) void prep_w1(const float* __restrict__ W1,
                                               _Float16* __restrict__ W1t) {
    const int i = blockIdx.x * 256 + threadIdx.x;     // 1024 * 32
    if (i >= HDIM * 32) return;
    const int n = i >> 5, k = i & 31;
    W1t[i] = (k < 8) ? (_Float16)W1[k * HDIM + n] : (_Float16)0.f;
}

__global__ __launch_bounds__(256) void prep_wt(const float* __restrict__ W,
                                               _Float16* __restrict__ Wt) {
    const int i = blockIdx.x * 256 + threadIdx.x;     // 1024 * 1024
    const int n = i >> 10, k = i & 1023;
    Wt[i] = (_Float16)W[(size_t)k * HDIM + n];        // Wt[n*1024 + k] = W[k,n]
}

// ---------------------------------------------------------------------------
// Fused MLP + hard constraints. One block = 64 rows.
__global__ __launch_bounds__(NTHR) void mlp_fused(
    const float*    __restrict__ x,
    const _Float16* __restrict__ W1t,
    const float*    __restrict__ b1,
    const _Float16* __restrict__ W2t,
    const float*    __restrict__ b2,
    const _Float16* __restrict__ W3t,
    const float*    __restrict__ b3,
    const float*    __restrict__ W4,
    const float*    __restrict__ b4,
    float*          __restrict__ out) {

    extern __shared__ char smem[];
    _Float16* xf  = (_Float16*)(smem + SMEM_XF);
    _Float16* h0  = (_Float16*)(smem + SMEM_H0);
    _Float16* h1  = (_Float16*)(smem + SMEM_H1);
    float*    red = (float*)(smem + SMEM_RED);

    const int rowBase = blockIdx.x * ROWS;
    const int tid = threadIdx.x;

    // Stage x -> f16, K padded 8 -> 32 with zeros.
    for (int i = tid; i < ROWS * 32; i += NTHR) {
        const int r = i >> 5, c = i & 31;
        xf[i] = (c < 8) ? (_Float16)x[(size_t)(rowBase + r) * 8 + c] : (_Float16)0.f;
    }
    __syncthreads();

    layer_wmma(xf, 32,   W1t, b1, h0);   // layer 1 (single K-step, padded)
    __syncthreads();
    layer_wmma(h0, HDIM, W2t, b2, h1);   // layer 2
    __syncthreads();
    layer_wmma(h1, HDIM, W3t, b3, h0);   // layer 3
    __syncthreads();

    // Layer 4: 64x1024 @ 1024x1. 4 threads per row, 256-K slices each.
    {
        const int row = tid & 63;
        const int q   = tid >> 6;                     // 0..3
        const _Float16* hr = h0 + row * HDIM + q * 256;
        const float*    wq = W4 + q * 256;
        float sum = 0.f;
        #pragma unroll 4
        for (int k = 0; k < 256; k += 8) {
            v8h hv = *(const v8h*)(hr + k);
            #pragma unroll
            for (int j = 0; j < 8; ++j)
                sum += (float)hv[j] * wq[k + j];
        }
        red[row * 4 + q] = sum;
    }
    __syncthreads();

    // Reduce + bias + hard physics constraints, write fp32 output.
    if (tid < ROWS) {
        const int grow = rowBase + tid;
        const float raw = red[tid * 4 + 0] + red[tid * 4 + 1] +
                          red[tid * 4 + 2] + red[tid * 4 + 3] + b4[0];

        const float* xr = x + (size_t)grow * 8;
        const float cement = xr[0];
        const float slag   = xr[1];
        const float fly    = xr[2];
        const float water  = xr[3];
        const float age    = fmaxf(xr[7], 1.0f);

        const bool  valid    = (cement > 0.0f) && (water > 0.0f);
        const float cem_safe = (cement > 0.0f) ? cement : 1.0f;
        const float w_c      = water / cem_safe;
        const float scm      = slag + fly;
        const float a_max    = fminf(0.95f, 0.95f - 0.15f * scm / fmaxf(cement + scm, 0.1f));
        const float k_hyd    = 0.01f / fmaxf(1.0f + w_c, 0.1f);
        const float alpha    = a_max * (1.0f - expf(-k_hyd * age));
        const float binder   = cement + scm;
        const float bsafe    = (binder > 0.0f) ? binder : 1.0f;
        const float wcsafe   = (w_c > 0.0f) ? w_c : 1.0f;
        const float gel      = alpha * (cement / bsafe) / wcsafe;
        const float g        = fminf(fmaxf(gel, 0.01f), 10.0f);
        const float max_str  = 50.0f * g * g * g;
        const float total    = cement + water + scm;
        const float comp     = cement / fmaxf(total, 1e-6f);
        const float cons_lim = 120.0f * comp;
        const float phys_lim = fminf(fmaxf(max_str, 5.0f), 120.0f);
        const float ub       = fminf(phys_lim, cons_lim);
        const float constr   = fminf(fmaxf(raw, 5.0f), ub);
        const bool  apply    = valid && (binder > 0.0f);

        out[grow] = apply ? constr : raw;
    }
}

// ---------------------------------------------------------------------------
extern "C" void kernel_launch(void* const* d_in, const int* in_sizes, int n_in,
                              void* d_out, int out_size, void* d_ws, size_t ws_size,
                              hipStream_t stream) {
    (void)in_sizes; (void)n_in; (void)out_size; (void)ws_size;

    const float* x  = (const float*)d_in[0];
    const float* W1 = (const float*)d_in[1];
    const float* b1 = (const float*)d_in[2];
    const float* W2 = (const float*)d_in[3];
    const float* b2 = (const float*)d_in[4];
    const float* W3 = (const float*)d_in[5];
    const float* b3 = (const float*)d_in[6];
    const float* W4 = (const float*)d_in[7];
    const float* b4 = (const float*)d_in[8];
    float* out = (float*)d_out;

    // Workspace: f16 transposed weights.
    char* ws = (char*)d_ws;
    _Float16* W1t = (_Float16*)(ws);                         // 1024 x 32  (64 KB)
    _Float16* W2t = (_Float16*)(ws + 65536);                 // 1024 x 1024 (2 MB)
    _Float16* W3t = (_Float16*)(ws + 65536 + 2097152);       // 1024 x 1024 (2 MB)

    prep_w1<<<(HDIM * 32 + 255) / 256, 256, 0, stream>>>(W1, W1t);
    prep_wt<<<(HDIM * HDIM) / 256, 256, 0, stream>>>(W2, W2t);
    prep_wt<<<(HDIM * HDIM) / 256, 256, 0, stream>>>(W3, W3t);

    // 261 KB dynamic LDS per workgroup (CDNA5 WGP: 320 KB).
    hipFuncSetAttribute((const void*)mlp_fused,
                        hipFuncAttributeMaxDynamicSharedMemorySize, SMEM_TOTAL);

    const int nBlocks = 32768 / ROWS;   // 512
    mlp_fused<<<nBlocks, NTHR, SMEM_TOTAL, stream>>>(
        x, W1t, b1, W2t, b2, W3t, b3, W4, b4, out);
}
